// TdownAndAtBlock_37701222924677
// MI455X (gfx1250) — matrix operand order repaired and checked by reading, not compiled, not run on hardware
//
#include <hip/hip_runtime.h>
#include <cstdint>
#include <cstddef>

// ---------------- problem constants ----------------
#define Bc      4
#define Np      8192
#define NPOINT  2048
#define Kc      16
#define CIN     128
#define CH      256
#define DM      512
#define KP      160            // (C_IN + 3)=131 padded up to a multiple of 32

#define BNP   (Bc * NPOINT)      // 8192 rows  (per-point)
#define BNPK  (Bc * NPOINT * Kc) // 131072 rows (per-(point,k))

typedef __bf16 bf16_t;
typedef __bf16 v16bf __attribute__((ext_vector_type(16)));
typedef __bf16 v8bf  __attribute__((ext_vector_type(8)));
typedef float  v8f   __attribute__((ext_vector_type(8)));
typedef unsigned int u32x4 __attribute__((ext_vector_type(4)));
typedef int          i32x8 __attribute__((ext_vector_type(8)));
typedef int          i32x4 __attribute__((ext_vector_type(4)));

#if defined(__has_builtin)
#  if __has_builtin(__builtin_amdgcn_tensor_load_to_lds) && \
      __has_builtin(__builtin_amdgcn_s_wait_tensorcnt)
#    define HAVE_TDM 1
#  endif
#endif
#ifndef HAVE_TDM
#  define HAVE_TDM 0
#endif

// ============================================================
// 1) Farthest Point Sampling: one block per batch.
// ============================================================
__global__ __launch_bounds__(1024)
void fps_kernel(const float* __restrict__ xyz, float* __restrict__ new_xyz) {
  const int b = blockIdx.x;
  const int tid = threadIdx.x;
  __shared__ float sd[1024];
  __shared__ int   si[1024];
  __shared__ float cent[3];

  float px[8], py[8], pz[8], dist[8];
#pragma unroll
  for (int j = 0; j < 8; ++j) {
    const int idx = j * 1024 + tid;
    const float* p = xyz + ((size_t)b * Np + idx) * 3;
    px[j] = p[0]; py[j] = p[1]; pz[j] = p[2];
    dist[j] = 1e10f;
  }
  int far = 0;
  for (int it = 0; it < NPOINT; ++it) {
    if (tid == 0) {
      const float* c = xyz + ((size_t)b * Np + far) * 3;
      const float cx = c[0], cy = c[1], cz = c[2];
      cent[0] = cx; cent[1] = cy; cent[2] = cz;
      float* o = new_xyz + ((size_t)b * NPOINT + it) * 3;
      o[0] = cx; o[1] = cy; o[2] = cz;
    }
    __syncthreads();
    const float cx = cent[0], cy = cent[1], cz = cent[2];
    float bestd = -1.f; int besti = tid;
#pragma unroll
    for (int j = 0; j < 8; ++j) {
      const float dx = px[j] - cx, dy = py[j] - cy, dz = pz[j] - cz;
      const float d = dx * dx + dy * dy + dz * dz;
      dist[j] = fminf(dist[j], d);
      if (dist[j] > bestd) { bestd = dist[j]; besti = j * 1024 + tid; }
    }
    sd[tid] = bestd; si[tid] = besti;
    __syncthreads();
    for (int s = 512; s > 0; s >>= 1) {
      if (tid < s && sd[tid + s] > sd[tid]) { sd[tid] = sd[tid + s]; si[tid] = si[tid + s]; }
      __syncthreads();
    }
    far = si[0];
  }
}

// ============================================================
// 2) kNN: one thread per query, register insertion-sort top-16.
// ============================================================
__global__ void knn_kernel(const float* __restrict__ query, const float* __restrict__ cand,
                           int* __restrict__ outIdx, int M, int NC) {
  const int g = blockIdx.x * blockDim.x + threadIdx.x;
  if (g >= Bc * M) return;
  const int b = g / M;
  const float qx = query[(size_t)g * 3 + 0];
  const float qy = query[(size_t)g * 3 + 1];
  const float qz = query[(size_t)g * 3 + 2];
  float bd[Kc]; int bi[Kc];
#pragma unroll
  for (int i = 0; i < Kc; ++i) { bd[i] = 3.0e38f; bi[i] = 0; }
  const float* cb = cand + (size_t)b * NC * 3;
  for (int n = 0; n < NC; ++n) {
    const float dx = cb[n * 3 + 0] - qx;
    const float dy = cb[n * 3 + 1] - qy;
    const float dz = cb[n * 3 + 2] - qz;
    const float d = dx * dx + dy * dy + dz * dz;
    if (d < bd[Kc - 1]) {
      bd[Kc - 1] = d; bi[Kc - 1] = n;
#pragma unroll
      for (int j = Kc - 1; j > 0; --j) {
        if (bd[j] < bd[j - 1]) {
          const float td = bd[j]; bd[j] = bd[j - 1]; bd[j - 1] = td;
          const int   ti = bi[j]; bi[j] = bi[j - 1]; bi[j - 1] = ti;
        }
      }
    }
  }
#pragma unroll
  for (int i = 0; i < Kc; ++i) outIdx[(size_t)g * Kc + i] = bi[i];
}

// ============================================================
// 3) Small utility kernels
// ============================================================
__global__ void zero_f32(float* p, int n) {
  const int i = blockIdx.x * blockDim.x + threadIdx.x;
  if (i < n) p[i] = 0.f;
}

__global__ void cvt_bf16(const float* __restrict__ s, bf16_t* __restrict__ d, int n) {
  const int i = blockIdx.x * blockDim.x + threadIdx.x;
  if (i < n) d[i] = (bf16_t)s[i];
}

__global__ void cvt_bf16_pad(const float* __restrict__ s, bf16_t* __restrict__ d,
                             int rows, int sc, int dc) {
  const int i = blockIdx.x * blockDim.x + threadIdx.x;
  if (i >= rows * dc) return;
  const int r = i / dc, c = i % dc;
  d[i] = (bf16_t)((c < sc) ? s[(size_t)r * sc + c] : 0.f);
}

__global__ void bn_fin(const float* __restrict__ ssum, const float* __restrict__ ssq,
                       const float* __restrict__ g, const float* __restrict__ be,
                       float* __restrict__ scale, float* __restrict__ shift, float count) {
  const int c = threadIdx.x;
  const float mean = ssum[c] / count;
  const float var  = ssq[c] / count - mean * mean;
  const float sc   = g[c] * rsqrtf(var + 1e-5f);
  scale[c] = sc;
  shift[c] = be[c] - mean * sc;
}

// ============================================================
// 4) LDS-tiled bf16 WMMA GEMM with TDM async staging.
//    Block tile 128(M) x 64(N), 8 waves, each wave a 32x32 micro-tile
//    (2x2 v_wmma_f32_16x16x32_bf16 accumulators).
//    A/B K-step tiles (rows x 32 bf16) staged in LDS with an 80-byte
//    row pitch (TDM pad: +4 DWORDs every 16 DWORDs -> conflict-free
//    16-row ds_load_b128 fragment reads), double-buffered,
//    synchronized with s_wait_tensorcnt + workgroup barrier.
// ============================================================
#define TM 128
#define TN 64
#define LDS_PITCH 40   // bf16 elems per staged row (80 bytes)

__device__ __forceinline__ uint32_t lds_off(const void* p) {
  // LDS aperture: low 32 bits of the flat address are the LDS byte offset
  return (uint32_t)(uintptr_t)p;
}

#if HAVE_TDM
// D# builder for a 2-D tile: rows x 32 bf16 (64B rows), moved as 8-byte units.
__device__ __forceinline__ void tdm_load_tile(const bf16_t* gsrc, uint32_t lds_byte,
                                              uint32_t rows, uint32_t kd_elems) {
  const uint64_t ga = (uint64_t)(uintptr_t)gsrc;
  u32x4 g0;
  g0.x = 1u;                                           // count=1, user mode, no gather
  g0.y = lds_byte;                                     // lds_addr
  g0.z = (uint32_t)ga;                                 // global_addr[31:0]
  g0.w = (uint32_t)((ga >> 32) & 0x01FFFFFFu) | 0x80000000u; // addr[56:32] | type=2
  const uint32_t tile_d0 = 8;                          // 8 x 8B = 64B per row
  const uint64_t stride  = (uint64_t)(kd_elems >> 2);  // row stride in 8B units
  i32x8 g1;
  g1[0] = (int)((3u << 16)            // data_size = 8B
              | (1u << 20)            // pad_enable
              | (3u << 22)            // pad_interval: every 16 DWORDs
              | (3u << 25));          // pad_amount: 4 DWORDs
  g1[1] = (int)0xFFFF0000u;                            // tensor_dim0 lo16 (huge)
  g1[2] = (int)((0xFFFFu << 16) | 0x7FFFu);            // dim0 hi16, dim1 lo16
  g1[3] = (int)((tile_d0 << 16) | 0x7FFFu);            // dim1 hi16, tile_dim0
  g1[4] = (int)(rows & 0xFFFFu);                       // tile_dim1, tile_dim2=0
  g1[5] = (int)(uint32_t)(stride & 0xFFFFFFFFu);       // dim0_stride lo32
  g1[6] = (int)(uint32_t)((stride >> 32) & 0xFFFFu);   // dim0_stride hi16
  g1[7] = 0;
  const i32x4 z4 = {0, 0, 0, 0};
  const i32x8 z8 = {0, 0, 0, 0, 0, 0, 0, 0};
  __builtin_amdgcn_tensor_load_to_lds(g0, g1, z4, z4, z8, 0);
}
#endif

template <int KD>
__global__ __launch_bounds__(256)
void wmma_gemm_bf16(const bf16_t* __restrict__ A, const bf16_t* __restrict__ W,
                    const float* __restrict__ bias,
                    const float* __restrict__ ascale, const float* __restrict__ ashift,
                    const float* __restrict__ resid,
                    float* __restrict__ Cf, bf16_t* __restrict__ Cb,
                    float* __restrict__ ssum, float* __restrict__ ssq,
                    int Nd, int relu) {
  __shared__ bf16_t As[2][TM * LDS_PITCH];   // 20 KB
  __shared__ bf16_t Bs[2][TN * LDS_PITCH];   // 10 KB

  const int lane = threadIdx.x & 31;
  const int wave = threadIdx.x >> 5;   // 0..7
  const int wm   = wave & 3;           // 4 waves along M
  const int wn   = wave >> 2;          // 2 waves along N
  const int half = lane >> 4;
  const int l15  = lane & 15;
  const int tileM = blockIdx.x * TM;
  const int tileN = blockIdx.y * TN;

  v8f acc[2][2];
#pragma unroll
  for (int ni = 0; ni < 2; ++ni) {
    const float bv = bias ? bias[tileN + wn * 32 + ni * 16 + l15] : 0.f;
#pragma unroll
    for (int mi = 0; mi < 2; ++mi)
#pragma unroll
      for (int r = 0; r < 8; ++r) acc[mi][ni][r] = bv;
  }

  auto stage = [&](int buf, int k) {
#if HAVE_TDM
    if (threadIdx.x < 32) {
      tdm_load_tile(A + (size_t)tileM * KD + k, lds_off(&As[buf][0]), TM, KD);
      tdm_load_tile(W + (size_t)tileN * KD + k, lds_off(&Bs[buf][0]), TN, KD);
    }
#else
    for (int c = threadIdx.x; c < TM * 4; c += 256) {
      const int r = c >> 2, ch = c & 3;
      *(v8bf*)(&As[buf][r * LDS_PITCH + ch * 8]) =
          *(const v8bf*)(A + (size_t)(tileM + r) * KD + k + ch * 8);
    }
    for (int c = threadIdx.x; c < TN * 4; c += 256) {
      const int r = c >> 2, ch = c & 3;
      *(v8bf*)(&Bs[buf][r * LDS_PITCH + ch * 8]) =
          *(const v8bf*)(W + (size_t)(tileN + r) * KD + k + ch * 8);
    }
#endif
  };

  constexpr int NSTEP = KD / 32;
  stage(0, 0);
#pragma unroll
  for (int ks = 0; ks < NSTEP; ++ks) {
    const int cur = ks & 1;
    const int k0 = ks * 32;
#if HAVE_TDM
    if (threadIdx.x < 32) __builtin_amdgcn_s_wait_tensorcnt(0);
#endif
    __syncthreads();
    if (ks + 1 < NSTEP) stage(cur ^ 1, k0 + 32);

    // ---- build fragments from LDS (80B pitch, all 16B aligned) ----
    v16bf af[2];
#pragma unroll
    for (int mi = 0; mi < 2; ++mi) {
      const bf16_t* p = &As[cur][(wm * 32 + mi * 16 + l15) * LDS_PITCH + half * 8];
      const v8bf lo = *(const v8bf*)(p);
      const v8bf hi = *(const v8bf*)(p + 16);
#pragma unroll
      for (int e = 0; e < 8; ++e) { af[mi][e] = lo[e]; af[mi][8 + e] = hi[e]; }
      if (ascale) {  // fused BN affine + ReLU on the A operand
#pragma unroll
        for (int e = 0; e < 16; ++e) {
          const int kc = k0 + half * 8 + e + ((e >= 8) ? 8 : 0);
          const float f = (float)af[mi][e] * ascale[kc] + ashift[kc];
          af[mi][e] = (bf16_t)(f > 0.f ? f : 0.f);
        }
      }
    }
    v16bf bw[2];
#pragma unroll
    for (int ni = 0; ni < 2; ++ni) {
      const bf16_t* p = &Bs[cur][(wn * 32 + ni * 16 + l15) * LDS_PITCH + half * 8];
      const v8bf lo = *(const v8bf*)(p);
      const v8bf hi = *(const v8bf*)(p + 16);
#pragma unroll
      for (int e = 0; e < 8; ++e) { bw[ni][e] = lo[e]; bw[ni][8 + e] = hi[e]; }
    }
#pragma unroll
    for (int mi = 0; mi < 2; ++mi)
#pragma unroll
      for (int ni = 0; ni < 2; ++ni)
        acc[mi][ni] = __builtin_amdgcn_wmma_f32_16x16x32_bf16(
            false, af[mi], false, bw[ni], (short)0, acc[mi][ni], false, false);
  }

  // ---- epilogue ----
#pragma unroll
  for (int ni = 0; ni < 2; ++ni) {
    const int colg = tileN + wn * 32 + ni * 16 + l15;
    float s = 0.f, q = 0.f;
#pragma unroll
    for (int mi = 0; mi < 2; ++mi) {
      v8f c = acc[mi][ni];
      if (relu) {
#pragma unroll
        for (int r = 0; r < 8; ++r) c[r] = c[r] > 0.f ? c[r] : 0.f;
      }
      const size_t base =
          (size_t)(tileM + wm * 32 + mi * 16 + half * 8) * Nd + colg;
      if (resid) {
#pragma unroll
        for (int r = 0; r < 8; ++r) c[r] += resid[base + (size_t)r * Nd];
      }
      if (Cf) {
#pragma unroll
        for (int r = 0; r < 8; ++r) Cf[base + (size_t)r * Nd] = c[r];
      }
      if (Cb) {
#pragma unroll
        for (int r = 0; r < 8; ++r) Cb[base + (size_t)r * Nd] = (bf16_t)c[r];
      }
      if (ssum) {
#pragma unroll
        for (int r = 0; r < 8; ++r) { s += c[r]; q += c[r] * c[r]; }
      }
    }
    if (ssum) { atomicAdd(&ssum[colg], s); atomicAdd(&ssq[colg], q); }
  }
}

// ============================================================
// 5) Gather / pointwise kernels
// ============================================================
__global__ __launch_bounds__(KP)
void build_grouped(const float* __restrict__ xyz, const float* __restrict__ pts,
                   const float* __restrict__ new_xyz, const int* __restrict__ knn,
                   bf16_t* __restrict__ Ab) {
  const int row = blockIdx.x;              // [0, BNPK)
  const int t   = threadIdx.x;             // [0, KP)
  const int b   = row / (NPOINT * Kc);
  const int m   = (row / Kc) % NPOINT;
  const int idx = knn[row];
  float v = 0.f;
  if (t < 3)
    v = xyz[((size_t)b * Np + idx) * 3 + t] - new_xyz[((size_t)b * NPOINT + m) * 3 + t];
  else if (t < 3 + CIN)
    v = pts[((size_t)b * Np + idx) * CIN + (t - 3)];
  Ab[(size_t)row * KP + t] = (bf16_t)v;
}

__global__ __launch_bounds__(CH)
void maxpool_kernel(const bf16_t* __restrict__ h2, const float* __restrict__ scale,
                    const float* __restrict__ shift, float* __restrict__ ff,
                    bf16_t* __restrict__ fb) {
  const int row = blockIdx.x;  // [0, BNP)
  const int c   = threadIdx.x;
  const float s = scale[c], t = shift[c];
  float mx = 0.f;
#pragma unroll
  for (int kk = 0; kk < Kc; ++kk) {
    float h = (float)h2[((size_t)row * Kc + kk) * CH + c] * s + t;
    h = h > 0.f ? h : 0.f;
    mx = fmaxf(mx, h);
  }
  ff[(size_t)row * CH + c] = mx;
  fb[(size_t)row * CH + c] = (bf16_t)mx;
}

__global__ __launch_bounds__(DM)
void pos1_kernel(const float* __restrict__ new_xyz, const int* __restrict__ knn2,
                 const float* __restrict__ dw1, const float* __restrict__ db1,
                 bf16_t* __restrict__ P1) {
  const int row = blockIdx.x;  // [0, BNPK)
  const int d   = threadIdx.x;
  const int b   = row / (NPOINT * Kc);
  const int m   = (row / Kc) % NPOINT;
  const int idx = knn2[row];
  const float* pm = new_xyz + ((size_t)b * NPOINT + m) * 3;
  const float* pk = new_xyz + ((size_t)b * NPOINT + idx) * 3;
  const float rx = pm[0] - pk[0], ry = pm[1] - pk[1], rz = pm[2] - pk[2];
  float s = dw1[d * 3 + 0] * rx + dw1[d * 3 + 1] * ry + dw1[d * 3 + 2] * rz + db1[d];
  P1[(size_t)row * DM + d] = (bf16_t)(s > 0.f ? s : 0.f);
}

__global__ __launch_bounds__(DM)
void a0_kernel(const bf16_t* __restrict__ qb, const bf16_t* __restrict__ kb,
               const bf16_t* __restrict__ pe, const int* __restrict__ knn2,
               bf16_t* __restrict__ dst) {
  const int row = blockIdx.x;  // [0, BNPK)
  const int d   = threadIdx.x;
  const int b   = row / (NPOINT * Kc);
  const int m   = (row / Kc) % NPOINT;
  const int idx = knn2[row];
  const float v = (float)qb[((size_t)b * NPOINT + m) * DM + d]
                - (float)kb[((size_t)b * NPOINT + idx) * DM + d]
                + (float)pe[(size_t)row * DM + d];
  dst[(size_t)row * DM + d] = (bf16_t)v;
}

__global__ __launch_bounds__(DM)
void attn_kernel(const bf16_t* __restrict__ a2, const bf16_t* __restrict__ vb,
                 const bf16_t* __restrict__ pe, const int* __restrict__ knn2,
                 bf16_t* __restrict__ res) {
  const int row = blockIdx.x;  // [0, BNP)
  const int d   = threadIdx.x;
  const int b   = row / NPOINT;
  __shared__ int sidx[Kc];
  if (d < Kc) sidx[d] = knn2[(size_t)row * Kc + d];
  __syncthreads();
  const float sc = 0.04419417382415922f;  // 1/sqrt(512)
  float vals[Kc]; float mx = -3.0e38f;
#pragma unroll
  for (int kk = 0; kk < Kc; ++kk) {
    const float a = (float)a2[((size_t)row * Kc + kk) * DM + d] * sc;
    vals[kk] = a; mx = fmaxf(mx, a);
  }
  float sum = 0.f;
#pragma unroll
  for (int kk = 0; kk < Kc; ++kk) { const float e = __expf(vals[kk] - mx); vals[kk] = e; sum += e; }
  const float inv = 1.f / sum;
  float acc = 0.f;
#pragma unroll
  for (int kk = 0; kk < Kc; ++kk) {
    const float vv = (float)vb[((size_t)b * NPOINT + sidx[kk]) * DM + d]
                   + (float)pe[((size_t)row * Kc + kk) * DM + d];
    acc += vals[kk] * inv * vv;
  }
  res[(size_t)row * DM + d] = (bf16_t)acc;
}

// ============================================================
// launcher
// ============================================================
extern "C" void kernel_launch(void* const* d_in, const int* in_sizes, int n_in,
                              void* d_out, int out_size, void* d_ws, size_t ws_size,
                              hipStream_t stream) {
  (void)in_sizes; (void)n_in; (void)out_size; (void)ws_size;

  const float* xyz    = (const float*)d_in[0];
  const float* pts    = (const float*)d_in[1];
  const float* td_w1  = (const float*)d_in[2];
  const float* td_b1  = (const float*)d_in[3];
  const float* td_g1  = (const float*)d_in[4];
  const float* td_be1 = (const float*)d_in[5];
  const float* td_w2  = (const float*)d_in[6];
  const float* td_b2  = (const float*)d_in[7];
  const float* td_g2  = (const float*)d_in[8];
  const float* td_be2 = (const float*)d_in[9];
  const float* fc1_w  = (const float*)d_in[10];
  const float* fc1_b  = (const float*)d_in[11];
  const float* fc2_w  = (const float*)d_in[12];
  const float* fc2_b  = (const float*)d_in[13];
  const float* wq     = (const float*)d_in[14];
  const float* wk     = (const float*)d_in[15];
  const float* wv     = (const float*)d_in[16];
  const float* dw1    = (const float*)d_in[17];
  const float* db1    = (const float*)d_in[18];
  const float* dw2    = (const float*)d_in[19];
  const float* db2    = (const float*)d_in[20];
  const float* gw1    = (const float*)d_in[21];
  const float* gb1    = (const float*)d_in[22];
  const float* gw2    = (const float*)d_in[23];
  const float* gb2    = (const float*)d_in[24];

  float* onew = (float*)d_out;                       // [B,NPOINT,3]
  float* oout = onew + (size_t)Bc * NPOINT * 3;      // [B,NPOINT,CH]

  char* wsb = (char*)d_ws;
  size_t off = 0;
  auto carve = [&](size_t bytes) -> char* {
    char* p = wsb + off;
    off = (off + bytes + 255) & ~(size_t)255;
    return p;
  };
  int*    knn1   = (int*)carve((size_t)BNPK * 4);
  int*    knn2   = (int*)carve((size_t)BNPK * 4);
  bf16_t* gA     = (bf16_t*)carve((size_t)BNPK * KP * 2);
  bf16_t* h1     = (bf16_t*)carve((size_t)BNPK * CH * 2);
  bf16_t* h2     = (bf16_t*)carve((size_t)BNPK * CH * 2);
  float*  stats  = (float*)carve(2048 * 4);
  float *ssum1 = stats,          *ssq1 = stats + CH;
  float *ssum2 = stats + 2 * CH, *ssq2 = stats + 3 * CH;
  float *scale1 = stats + 4 * CH, *shift1 = stats + 5 * CH;
  float *scale2 = stats + 6 * CH, *shift2 = stats + 7 * CH;
  float*  feat_f = (float*)carve((size_t)BNP * CH * 4);
  bf16_t* feat_b = (bf16_t*)carve((size_t)BNP * CH * 2);
  bf16_t* x_b    = (bf16_t*)carve((size_t)BNP * DM * 2);
  bf16_t* q_b    = (bf16_t*)carve((size_t)BNP * DM * 2);
  bf16_t* k_b    = (bf16_t*)carve((size_t)BNP * DM * 2);
  bf16_t* v_b    = (bf16_t*)carve((size_t)BNP * DM * 2);
  bf16_t* res_b  = (bf16_t*)carve((size_t)BNP * DM * 2);
  bf16_t* P1     = (bf16_t*)carve((size_t)BNPK * DM * 2);  // pos1 -> a1
  bf16_t* P2     = (bf16_t*)carve((size_t)BNPK * DM * 2);  // pos_enc (long-lived)
  bf16_t* P3     = (bf16_t*)carve((size_t)BNPK * DM * 2);  // a0 -> a2
  bf16_t* w1b    = (bf16_t*)carve((size_t)CH * KP * 2);
  bf16_t* w2b    = (bf16_t*)carve((size_t)CH * CH * 2);
  bf16_t* fc1b   = (bf16_t*)carve((size_t)DM * CH * 2);
  bf16_t* wqb    = (bf16_t*)carve((size_t)DM * DM * 2);
  bf16_t* wkb    = (bf16_t*)carve((size_t)DM * DM * 2);
  bf16_t* wvb    = (bf16_t*)carve((size_t)DM * DM * 2);
  bf16_t* dw2b   = (bf16_t*)carve((size_t)DM * DM * 2);
  bf16_t* gw1b   = (bf16_t*)carve((size_t)DM * DM * 2);
  bf16_t* gw2b   = (bf16_t*)carve((size_t)DM * DM * 2);
  bf16_t* fc2b   = (bf16_t*)carve((size_t)CH * DM * 2);

  // ---- sampling / neighbor structure ----
  fps_kernel<<<Bc, 1024, 0, stream>>>(xyz, onew);
  knn_kernel<<<(Bc * NPOINT + 127) / 128, 128, 0, stream>>>(onew, xyz,  knn1, NPOINT, Np);
  knn_kernel<<<(Bc * NPOINT + 127) / 128, 128, 0, stream>>>(onew, onew, knn2, NPOINT, NPOINT);

  // ---- weights -> bf16, BN stat accumulators -> 0 ----
  zero_f32<<<2, 1024, 0, stream>>>(stats, 2048);
  cvt_bf16_pad<<<(CH * KP + 255) / 256, 256, 0, stream>>>(td_w1, w1b, CH, CIN + 3, KP);
  cvt_bf16<<<(CH * CH + 255) / 256, 256, 0, stream>>>(td_w2, w2b, CH * CH);
  cvt_bf16<<<(DM * CH + 255) / 256, 256, 0, stream>>>(fc1_w, fc1b, DM * CH);
  cvt_bf16<<<(DM * DM + 255) / 256, 256, 0, stream>>>(wq,  wqb,  DM * DM);
  cvt_bf16<<<(DM * DM + 255) / 256, 256, 0, stream>>>(wk,  wkb,  DM * DM);
  cvt_bf16<<<(DM * DM + 255) / 256, 256, 0, stream>>>(wv,  wvb,  DM * DM);
  cvt_bf16<<<(DM * DM + 255) / 256, 256, 0, stream>>>(dw2, dw2b, DM * DM);
  cvt_bf16<<<(DM * DM + 255) / 256, 256, 0, stream>>>(gw1, gw1b, DM * DM);
  cvt_bf16<<<(DM * DM + 255) / 256, 256, 0, stream>>>(gw2, gw2b, DM * DM);
  cvt_bf16<<<(CH * DM + 255) / 256, 256, 0, stream>>>(fc2_w, fc2b, CH * DM);

  // ---- TransitionDown ----
  build_grouped<<<BNPK, KP, 0, stream>>>(xyz, pts, onew, knn1, gA);
  const dim3 blk(256);
  const dim3 gTD(BNPK / TM, CH / TN);
  wmma_gemm_bf16<KP><<<gTD, blk, 0, stream>>>(gA, w1b, td_b1, nullptr, nullptr, nullptr,
                                              nullptr, h1, ssum1, ssq1, CH, 0);
  bn_fin<<<1, CH, 0, stream>>>(ssum1, ssq1, td_g1, td_be1, scale1, shift1, (float)BNPK);
  wmma_gemm_bf16<CH><<<gTD, blk, 0, stream>>>(h1, w2b, td_b2, scale1, shift1, nullptr,
                                              nullptr, h2, ssum2, ssq2, CH, 0);
  bn_fin<<<1, CH, 0, stream>>>(ssum2, ssq2, td_g2, td_be2, scale2, shift2, (float)BNPK);
  maxpool_kernel<<<BNP, CH, 0, stream>>>(h2, scale2, shift2, feat_f, feat_b);

  // ---- Transformer projections ----
  const dim3 gX(BNP / TM, DM / TN);
  wmma_gemm_bf16<CH><<<gX, blk, 0, stream>>>(feat_b, fc1b, fc1_b, nullptr, nullptr, nullptr,
                                             nullptr, x_b, nullptr, nullptr, DM, 0);
  wmma_gemm_bf16<DM><<<gX, blk, 0, stream>>>(x_b, wqb, nullptr, nullptr, nullptr, nullptr,
                                             nullptr, q_b, nullptr, nullptr, DM, 0);
  wmma_gemm_bf16<DM><<<gX, blk, 0, stream>>>(x_b, wkb, nullptr, nullptr, nullptr, nullptr,
                                             nullptr, k_b, nullptr, nullptr, DM, 0);
  wmma_gemm_bf16<DM><<<gX, blk, 0, stream>>>(x_b, wvb, nullptr, nullptr, nullptr, nullptr,
                                             nullptr, v_b, nullptr, nullptr, DM, 0);

  // ---- positional encoding + attention MLP (FLOP-dominant GEMMs) ----
  pos1_kernel<<<BNPK, DM, 0, stream>>>(onew, knn2, dw1, db1, P1);
  const dim3 gP(BNPK / TM, DM / TN);
  wmma_gemm_bf16<DM><<<gP, blk, 0, stream>>>(P1, dw2b, db2, nullptr, nullptr, nullptr,
                                             nullptr, P2, nullptr, nullptr, DM, 0);   // pos_enc
  a0_kernel<<<BNPK, DM, 0, stream>>>(q_b, k_b, P2, knn2, P3);                          // a0
  wmma_gemm_bf16<DM><<<gP, blk, 0, stream>>>(P3, gw1b, gb1, nullptr, nullptr, nullptr,
                                             nullptr, P1, nullptr, nullptr, DM, 1);   // a1=relu
  wmma_gemm_bf16<DM><<<gP, blk, 0, stream>>>(P1, gw2b, gb2, nullptr, nullptr, nullptr,
                                             nullptr, P3, nullptr, nullptr, DM, 0);   // a2

  // ---- softmax over K + vector attention + output projection ----
  attn_kernel<<<BNP, DM, 0, stream>>>(P3, v_b, P2, knn2, res_b);
  const dim3 gO(BNP / TM, CH / TN);
  wmma_gemm_bf16<DM><<<gO, blk, 0, stream>>>(res_b, fc2b, fc2_b, nullptr, nullptr, feat_f,
                                             oout, nullptr, nullptr, nullptr, CH, 0);
}
